// HMMTransducer_274877907284
// MI455X (gfx1250) — matrix operand order repaired and checked by reading, not compiled, not run on hardware
//
#include <hip/hip_runtime.h>
#include <math.h>

// ---------------------------------------------------------------------------
// HMM transducer for MI455X (gfx1250, wave32).
// All GEMMs: f16-input / f32-accumulate V_WMMA_F32_16X16X32_F16.
// Weights packed once into WMMA B-fragment layout, activations into A-fragment
// layout. Large GEMMs use 128x128 block tiles staged through LDS with double
// buffering (8x B reuse, 2x A reuse) -- the L2-bandwidth fix the roofline
// demands. Tiny M=32 recurrent GEMMs use a direct-global WMMA kernel.
// ---------------------------------------------------------------------------

typedef __attribute__((ext_vector_type(16))) _Float16 v16h;
typedef __attribute__((ext_vector_type(8)))  float    v8f;

// Problem constants (match setup_inputs()).
#define SS   64
#define BB   32
#define TT   48
#define EE   256
#define HH   512     // Hs == Ht
#define H4   2048    // 4*H
#define VV   256     // target vocab
#define OD   1536    // out_dim = Ht + 2*Hs

#define WMMA_F16(a, b, c) \
  __builtin_amdgcn_wmma_f32_16x16x32_f16(false, (a), false, (b), (short)0, (c), false, false)

// A-fragment element mapping (ISA 7.12.2, 16-bit A 16x32):
// lane = half*16 + (m%16); e<8 -> k_in = half*8+e ; e>=8 -> k_in = 16+half*8+(e-8)
__device__ __host__ __forceinline__ int a_kin(int half, int e) {
  return (e < 8) ? (half * 8 + e) : (16 + half * 8 + (e - 8));
}

// ---------------- weight packing: W[N,K] row-major -> B-fragment tiles ------
// tile t = kTile*(N/16)+nTile, 512 halves/tile; (lane,e): n = nTile*16+lane%16,
// k = kTile*32 + a_kin(lane/16, e).
__global__ void pack_weight(const float* __restrict__ W, _Float16* __restrict__ Bp,
                            int N, int K) {
  size_t total = (size_t)N * K;
  size_t i = (size_t)blockIdx.x * blockDim.x + threadIdx.x;
  if (i >= total) return;
  size_t tile = i >> 9;
  int within = (int)(i & 511);
  int lane = within >> 4;
  int e    = within & 15;
  int NT = N >> 4;
  int kTile = (int)(tile / NT);
  int nTile = (int)(tile % NT);
  int n = nTile * 16 + (lane & 15);
  int k = kTile * 32 + a_kin(lane >> 4, e);
  Bp[i] = (_Float16)W[(size_t)n * K + k];
}

// ---------------- activation packing: A[M,K] fp32 -> A-fragment tiles -------
// tile t = mTile*(K/32)+kTile (K-tile-major => contiguous stream per wave).
__global__ void pack_act(const float* __restrict__ A, _Float16* __restrict__ Ap,
                         int M, int K) {
  size_t total = (size_t)M * K;
  size_t i = (size_t)blockIdx.x * blockDim.x + threadIdx.x;
  if (i >= total) return;
  size_t tile = i >> 9;
  int within = (int)(i & 511);
  int lane = within >> 4;
  int e    = within & 15;
  int KT = K >> 5;
  int mTile = (int)(tile / KT);
  int kTile = (int)(tile % KT);
  int m = mTile * 16 + (lane & 15);
  int k = kTile * 32 + a_kin(lane >> 4, e);
  Ap[i] = (_Float16)A[(size_t)m * K + k];
}

// --------- LDS-tiled GEMM: C[M,N] = act(Ap @ Bp + bias) ---------------------
// Block = 256 threads = 8 waves (4 along M x 2 along N), block tile 128x128.
// Per 32-deep K-chunk: stage 8KB A panel + 8KB B panel into LDS (fragment
// layout, pure float4 copies), double-buffered. Each wave: 2 A frags x 4 B
// frags -> 8 WMMAs per chunk. Requires M%128==0, N%128==0, K%32==0.
__global__ void gemm_wmma_lds(const _Float16* __restrict__ Ap, const _Float16* __restrict__ Bp,
                              const float* __restrict__ bias, float* __restrict__ C,
                              int M, int N, int K, int act) {
  __shared__ _Float16 lsA[2][8 * 512];
  __shared__ _Float16 lsB[2][8 * 512];
  int tid  = threadIdx.x;
  int lane = tid & 31;
  int wave = tid >> 5;
  int waveM = wave >> 1;              // 0..3
  int waveN = wave & 1;               // 0..1
  int NT = N >> 4;
  int KT = K >> 5;
  int nb = N >> 7;
  int bM = blockIdx.x / nb;
  int bN = blockIdx.x % nb;

  int jA = tid >> 5;                  // which of the 8 A tiles this wave stages
  int wA = tid & 31;

  float4 ra0, ra1, rb0, rb1;
  auto loadG = [&](int kt) {
    const float4* pa = (const float4*)Ap + ((size_t)(bM * 8 + jA) * KT + kt) * 64;
    ra0 = pa[wA];
    ra1 = pa[wA + 32];
    const float4* pb = (const float4*)Bp + ((size_t)kt * NT + bN * 8) * 64;
    rb0 = pb[tid];
    rb1 = pb[tid + 256];
  };
  auto storeS = [&](int buf) {
    ((float4*)&lsA[buf][jA * 512])[wA]      = ra0;
    ((float4*)&lsA[buf][jA * 512])[wA + 32] = ra1;
    ((float4*)lsB[buf])[tid]       = rb0;
    ((float4*)lsB[buf])[tid + 256] = rb1;
  };

  v8f acc[2][4] = {};

  loadG(0);
  storeS(0);
  __syncthreads();

  for (int kt = 0; kt < KT; ++kt) {
    int buf = kt & 1;
    if (kt + 1 < KT) loadG(kt + 1);          // global prefetch overlaps WMMAs

    v16h a0 = *(const v16h*)&lsA[buf][(waveM * 2 + 0) * 512 + lane * 16];
    v16h a1 = *(const v16h*)&lsA[buf][(waveM * 2 + 1) * 512 + lane * 16];
#pragma unroll
    for (int q = 0; q < 4; ++q) {
      v16h bf = *(const v16h*)&lsB[buf][(waveN * 4 + q) * 512 + lane * 16];
      acc[0][q] = WMMA_F16(a0, bf, acc[0][q]);
      acc[1][q] = WMMA_F16(a1, bf, acc[1][q]);
    }
    if (kt + 1 < KT) {
      storeS(buf ^ 1);                       // other buffer: no conflict with readers
      __syncthreads();
    }
  }

  int half = lane >> 4, l16 = lane & 15;     // C layout: VGPR r -> M = half*8 + r
#pragma unroll
  for (int i2 = 0; i2 < 2; ++i2) {
    int mT = bM * 8 + waveM * 2 + i2;
#pragma unroll
    for (int q = 0; q < 4; ++q) {
      int n = (bN * 8 + waveN * 4 + q) * 16 + l16;
      float bv = bias ? bias[n] : 0.0f;
#pragma unroll
      for (int r = 0; r < 8; ++r) {
        float v = acc[i2][q][r] + bv;
        if (act == 1) v = tanhf(v);
        C[(size_t)(mT * 16 + half * 8 + r) * N + n] = v;
      }
    }
  }
}

// --------- fp32-A GEMM (inline cvt) for the tiny M=32 recurrent steps ------
__global__ void gemm_wmma(const float* __restrict__ A, const _Float16* __restrict__ Bp,
                          const float* __restrict__ bias, float* __restrict__ C,
                          int M, int N, int K, int act) {
  int lane = threadIdx.x & 31;
  int wave = threadIdx.x >> 5;
  int NT4  = N >> 6;
  int tiles = (M >> 4) * NT4;
  int gt = blockIdx.x * 8 + wave;
  if (gt >= tiles) return;
  int tM  = gt / NT4;
  int tN4 = gt % NT4;
  int half = lane >> 4;
  int l16  = lane & 15;

  v8f acc0 = {}, acc1 = {}, acc2 = {}, acc3 = {};
  const float* arow = A + (size_t)(tM * 16 + l16) * K;
  int NT = N >> 4;

  for (int k0 = 0; k0 < K; k0 += 32) {
    v16h a;
    const float* ap = arow + k0 + half * 8;
#pragma unroll
    for (int e = 0; e < 8; ++e) a[e]     = (_Float16)ap[e];
#pragma unroll
    for (int e = 0; e < 8; ++e) a[8 + e] = (_Float16)ap[16 + e];

    size_t tbase = ((size_t)(k0 >> 5) * NT + tN4 * 4) * 512 + (size_t)lane * 16;
    v16h b0 = *(const v16h*)(Bp + tbase);
    v16h b1 = *(const v16h*)(Bp + tbase + 512);
    v16h b2 = *(const v16h*)(Bp + tbase + 1024);
    v16h b3 = *(const v16h*)(Bp + tbase + 1536);

    acc0 = WMMA_F16(a, b0, acc0);
    acc1 = WMMA_F16(a, b1, acc1);
    acc2 = WMMA_F16(a, b2, acc2);
    acc3 = WMMA_F16(a, b3, acc3);
  }

  int mBase = tM * 16 + half * 8;
#pragma unroll
  for (int q = 0; q < 4; ++q) {
    v8f acc = (q == 0) ? acc0 : (q == 1) ? acc1 : (q == 2) ? acc2 : acc3;
    int n = (tN4 * 4 + q) * 16 + l16;
    float bv = bias ? bias[n] : 0.0f;
#pragma unroll
    for (int r = 0; r < 8; ++r) {
      float v = acc[r] + bv;
      if (act == 1) v = tanhf(v);
      C[(size_t)(mBase + r) * N + n] = v;
    }
  }
}

// ---------------- small helper kernels --------------------------------------
__global__ void fill_zero(float* __restrict__ p, int n) {
  int i = blockIdx.x * blockDim.x + threadIdx.x;
  if (i < n) p[i] = 0.0f;
}

__global__ void gather_rows(const float* __restrict__ emb, const int* __restrict__ idx,
                            float* __restrict__ out, int rows, int E) {
  size_t i = (size_t)blockIdx.x * blockDim.x + threadIdx.x;
  if (i >= (size_t)rows * E) return;
  int r = (int)(i / E);
  int e = (int)(i - (size_t)r * E);
  out[i] = emb[(size_t)idx[r] * E + e];
}

__device__ __forceinline__ float sigf(float x) { return 1.0f / (1.0f + expf(-x)); }

// gates order i,f,g,o (torch LSTMCell). preX/preH are [32, 4H] row blocks.
__global__ void lstm_cell(const float* __restrict__ preX, const float* __restrict__ preH,
                          const float* __restrict__ bias, float* __restrict__ h,
                          float* __restrict__ c, float* __restrict__ hout, int houtStride) {
  int idx = blockIdx.x * blockDim.x + threadIdx.x;
  if (idx >= BB * HH) return;
  int b = idx / HH, j = idx - b * HH;
  const float* px = preX + (size_t)b * H4;
  const float* ph = preH + (size_t)b * H4;
  float gi = px[j]          + ph[j]          + bias[j];
  float gf = px[HH + j]     + ph[HH + j]     + bias[HH + j];
  float gg = px[2*HH + j]   + ph[2*HH + j]   + bias[2*HH + j];
  float go = px[3*HH + j]   + ph[3*HH + j]   + bias[3*HH + j];
  float cc = sigf(gf) * c[idx] + sigf(gi) * tanhf(gg);
  float hh = sigf(go) * tanhf(cc);
  c[idx] = cc; h[idx] = hh;
  if (hout) hout[(size_t)b * houtStride + j] = hh;
}

// score[b,s] = dot(scale[s,b,:], h1[b,:])
__global__ void attn_score(const float* __restrict__ scale, const float* __restrict__ h1,
                           float* __restrict__ score) {
  int i = blockIdx.x * blockDim.x + threadIdx.x;
  if (i >= BB * SS) return;
  int b = i / SS, s = i - b * SS;
  const float* sr = scale + (size_t)(s * BB + b) * HH;
  const float* hr = h1 + (size_t)b * HH;
  float acc = 0.0f;
#pragma unroll 8
  for (int k = 0; k < HH; ++k) acc = fmaf(sr[k], hr[k], acc);
  score[i] = acc;
}

// masked softmax + eps + renorm + log -> ltr_t[b,s]. One wave per batch row.
__global__ void attn_norm(const float* __restrict__ score, const float* __restrict__ mask,
                          float* __restrict__ ltr_t) {
  int b = blockIdx.x;
  int lane = threadIdx.x;                       // handles s=lane and s=lane+32
  float v0 = score[b * SS + lane];
  float v1 = score[b * SS + lane + 32];
  float mx = fmaxf(v0, v1);
  for (int o = 16; o > 0; o >>= 1) mx = fmaxf(mx, __shfl_xor(mx, o, 32));
  float e0 = expf(v0 - mx), e1 = expf(v1 - mx);
  float sm = e0 + e1;
  for (int o = 16; o > 0; o >>= 1) sm += __shfl_xor(sm, o, 32);
  float a0 = e0 / sm * mask[lane * BB + b]        + 1e-7f;
  float a1 = e1 / sm * mask[(lane + 32) * BB + b] + 1e-7f;
  float s2 = a0 + a1;
  for (int o = 16; o > 0; o >>= 1) s2 += __shfl_xor(s2, o, 32);
  float ls2 = logf(s2);
  ltr_t[b * SS + lane]      = logf(a0) - ls2;
  ltr_t[b * SS + lane + 32] = logf(a1) - ls2;
}

// ---- packed ctxA builders (ctxAp is a packed-A f16 buffer, M=2048, K=1536) -
__device__ __forceinline__ size_t packedA_idx(int m, int k, int KT) {
  int mTile = m >> 4, kTile = k >> 5;
  int kin = k & 31;
  int half, e;
  if (kin < 16) { half = kin >> 3; e = kin & 7; }
  else          { half = (kin - 16) >> 3; e = 8 + ((kin - 16) & 7); }
  int lane = half * 16 + (m & 15);
  return ((size_t)mTile * KT + kTile) * 512 + (size_t)lane * 16 + e;
}

// cols [512,1536) = enc[s,b,:]  (written once; m = b*64+s)
__global__ void ctx_fill_enc_packed(const float* __restrict__ enc, _Float16* __restrict__ ctxAp) {
  size_t i = (size_t)blockIdx.x * blockDim.x + threadIdx.x;
  if (i >= (size_t)BB * SS * 1024) return;
  int m  = (int)(i >> 10);
  int k2 = (int)(i & 1023);
  int b = m / SS, s = m - b * SS;
  ctxAp[packedA_idx(m, 512 + k2, OD >> 5)] = (_Float16)enc[(size_t)(s * BB + b) * 1024 + k2];
}

// cols [0,512) = broadcast h1[b,:]  (per decoder step)
__global__ void ctx_fill_h_packed(const float* __restrict__ h1, _Float16* __restrict__ ctxAp) {
  size_t i = (size_t)blockIdx.x * blockDim.x + threadIdx.x;
  if (i >= (size_t)BB * SS * HH) return;
  int m = (int)(i >> 9);
  int k = (int)(i & 511);
  int b = m / SS;
  ctxAp[packedA_idx(m, k, OD >> 5)] = (_Float16)h1[(size_t)b * HH + k];
}

// log_softmax over V=256; one wave per row; write directly into d_out slice.
__global__ void log_softmax_rows(const float* __restrict__ logits, float* __restrict__ out,
                                 int rows) {
  int lane = threadIdx.x & 31;
  int wave = threadIdx.x >> 5;
  int row = blockIdx.x * 8 + wave;
  if (row >= rows) return;
  float x[8];
  float mx = -3.4e38f;
#pragma unroll
  for (int j = 0; j < 8; ++j) { x[j] = logits[(size_t)row * VV + j * 32 + lane]; mx = fmaxf(mx, x[j]); }
  for (int o = 16; o > 0; o >>= 1) mx = fmaxf(mx, __shfl_xor(mx, o, 32));
  float sm = 0.0f;
#pragma unroll
  for (int j = 0; j < 8; ++j) sm += expf(x[j] - mx);
  for (int o = 16; o > 0; o >>= 1) sm += __shfl_xor(sm, o, 32);
  float l = mx + logf(sm);
#pragma unroll
  for (int j = 0; j < 8; ++j) out[(size_t)row * VV + j * 32 + lane] = x[j] - l;
}

__global__ void copy_initial(const float* __restrict__ ltr0, float* __restrict__ out) {
  int i = blockIdx.x * blockDim.x + threadIdx.x;
  if (i < BB * SS) out[i] = ltr0[i];
}

// transition[tt][b][i][j] = ltr[tt+1][b][j]  (rows identical, torch expand)
__global__ void transition_bcast(const float* __restrict__ ltr, float* __restrict__ out) {
  size_t i = (size_t)blockIdx.x * blockDim.x + threadIdx.x;
  size_t total = (size_t)(TT - 2) * BB * SS * SS;
  if (i >= total) return;
  int j  = (int)(i & 63);
  size_t i3 = i >> 12;          // drop j and i (row index, unused)
  int b  = (int)(i3 & 31);
  int tt = (int)(i3 >> 5);
  out[i] = ltr[(size_t)(tt + 1) * BB * SS + b * SS + j];
}

// ---------------------------------------------------------------------------
extern "C" void kernel_launch(void* const* d_in, const int* in_sizes, int n_in,
                              void* d_out, int out_size, void* d_ws, size_t ws_size,
                              hipStream_t stream) {
  (void)in_sizes; (void)n_in; (void)out_size; (void)ws_size;
  const float* src_emb  = (const float*)d_in[0];
  const float* trg_emb  = (const float*)d_in[1];
  const float* Wih[4] = {(const float*)d_in[2],  (const float*)d_in[5],
                         (const float*)d_in[8],  (const float*)d_in[11]};   // l0f,l0b,l1f,l1b
  const float* Whh[4] = {(const float*)d_in[3],  (const float*)d_in[6],
                         (const float*)d_in[9],  (const float*)d_in[12]};
  const float* bE[4]  = {(const float*)d_in[4],  (const float*)d_in[7],
                         (const float*)d_in[10], (const float*)d_in[13]};
  const float* dWih0 = (const float*)d_in[14]; const float* dWhh0 = (const float*)d_in[15];
  const float* dB0   = (const float*)d_in[16];
  const float* dWih1 = (const float*)d_in[17]; const float* dWhh1 = (const float*)d_in[18];
  const float* dB1   = (const float*)d_in[19];
  const float* Wsc = (const float*)d_in[20]; const float* bsc = (const float*)d_in[21];
  const float* Wlo = (const float*)d_in[22]; const float* blo = (const float*)d_in[23];
  const float* Wfo = (const float*)d_in[24]; const float* bfo = (const float*)d_in[25];
  const float* src_mask = (const float*)d_in[26];
  const int* src_batch  = (const int*)d_in[27];
  const int* trg_batch  = (const int*)d_in[28];
  float* out = (float*)d_out;

  const int MDEC = 1536;                 // decoder pre-GEMM M padded 1504 -> 1536

  // ---- workspace carve-out ----
  char* ws = (char*)d_ws;
  size_t off = 0;
  auto alloc = [&](size_t bytes) -> void* {
    void* p = ws + off;
    off = (off + bytes + 255) & ~(size_t)255;
    return p;
  };
  auto allocF = [&](size_t n) -> float*      { return (float*)alloc(n * 4); };
  auto allocH = [&](size_t n) -> _Float16*   { return (_Float16*)alloc(n * 2); };

  _Float16* pIh[4], *pHh[4];
  pIh[0] = allocH((size_t)H4 * EE);   pIh[1] = allocH((size_t)H4 * EE);
  pIh[2] = allocH((size_t)H4 * 1024); pIh[3] = allocH((size_t)H4 * 1024);
  for (int d = 0; d < 4; ++d) pHh[d] = allocH((size_t)H4 * HH);
  _Float16* pDih0 = allocH((size_t)H4 * EE);
  _Float16* pDhh0 = allocH((size_t)H4 * HH);
  _Float16* pDih1 = allocH((size_t)H4 * HH);
  _Float16* pDhh1 = allocH((size_t)H4 * HH);
  _Float16* pSc   = allocH((size_t)HH * 1024);
  _Float16* pLo   = allocH((size_t)OD * OD);
  _Float16* pFo   = allocH((size_t)VV * OD);

  _Float16* ctxAp = allocH((size_t)BB * SS * OD);   // packed-A ctx (persistent)
  _Float16* apScr = allocH((size_t)BB * SS * OD);   // packed-A scratch (sequential reuse)

  float* xsrc   = allocF((size_t)SS * BB * EE);
  float* xdec   = allocF((size_t)MDEC * EE);        // padded to 1536 rows
  float* preBuf = allocF((size_t)SS * BB * H4);     // reused by all big pre-GEMMs
  float* h0buf  = allocF((size_t)SS * BB * 1024);
  float* encbuf = allocF((size_t)SS * BB * 1024);
  float* scaleB = allocF((size_t)SS * BB * HH);
  float* ctx1   = allocF((size_t)BB * SS * OD);
  float* logits = allocF((size_t)BB * SS * VV);
  float* ltr    = allocF((size_t)(TT - 1) * BB * SS);
  float* score  = allocF((size_t)BB * SS);
  float* hg0    = allocF((size_t)BB * H4);
  float* xg1    = allocF((size_t)BB * H4);
  float* hg1    = allocF((size_t)BB * H4);
  float* hSt    = allocF((size_t)BB * HH);  float* cSt = allocF((size_t)BB * HH);
  float* h0St   = allocF((size_t)BB * HH);  float* c0St = allocF((size_t)BB * HH);
  float* h1St   = allocF((size_t)BB * HH);  float* c1St = allocF((size_t)BB * HH);

  auto packW = [&](const float* W, _Float16* Bp, int N, int K) {
    size_t total = (size_t)N * K;
    pack_weight<<<(unsigned)((total + 255) / 256), 256, 0, stream>>>(W, Bp, N, K);
  };
  auto packA = [&](const float* A, _Float16* Ap, int M, int K) {
    size_t total = (size_t)M * K;
    pack_act<<<(unsigned)((total + 255) / 256), 256, 0, stream>>>(A, Ap, M, K);
  };
  auto gemmLDS = [&](const _Float16* Ap, const _Float16* Bp, const float* bias, float* C,
                     int M, int N, int K, int act) {
    int blocks = (M / 128) * (N / 128);
    gemm_wmma_lds<<<blocks, 256, 0, stream>>>(Ap, Bp, bias, C, M, N, K, act);
  };
  auto gemmF = [&](const float* A, const _Float16* Bp, const float* bias, float* C,
                   int M, int N, int K, int act) {
    int tiles = (M / 16) * (N / 64);
    gemm_wmma<<<(tiles + 7) / 8, 256, 0, stream>>>(A, Bp, bias, C, M, N, K, act);
  };
  auto zero = [&](float* p, int n) {
    fill_zero<<<(n + 255) / 256, 256, 0, stream>>>(p, n);
  };

  // ---- 1. pack all weights to f16 WMMA B-fragments ----
  packW(Wih[0], pIh[0], H4, EE);   packW(Wih[1], pIh[1], H4, EE);
  packW(Wih[2], pIh[2], H4, 1024); packW(Wih[3], pIh[3], H4, 1024);
  for (int d = 0; d < 4; ++d) packW(Whh[d], pHh[d], H4, HH);
  packW(dWih0, pDih0, H4, EE);  packW(dWhh0, pDhh0, H4, HH);
  packW(dWih1, pDih1, H4, HH);  packW(dWhh1, pDhh1, H4, HH);
  packW(Wsc, pSc, HH, 1024);
  packW(Wlo, pLo, OD, OD);
  packW(Wfo, pFo, VV, OD);

  // ---- 2. embedding gathers (xdec zero-padded to 1536 rows) ----
  {
    size_t n1 = (size_t)SS * BB * EE;
    gather_rows<<<(unsigned)((n1 + 255) / 256), 256, 0, stream>>>(src_emb, src_batch, xsrc, SS * BB, EE);
    zero(xdec, MDEC * EE);
    size_t n2 = (size_t)(TT - 1) * BB * EE;
    gather_rows<<<(unsigned)((n2 + 255) / 256), 256, 0, stream>>>(trg_emb, trg_batch, xdec, (TT - 1) * BB, EE);
  }

  // ---- 3. encoder: 2-layer biLSTM ----
  const float* layerIn[2] = {xsrc, h0buf};
  float* layerOut[2] = {h0buf, encbuf};
  int layerK[2] = {EE, 1024};
  for (int layer = 0; layer < 2; ++layer) {
    packA(layerIn[layer], apScr, SS * BB, layerK[layer]);      // shared by both dirs
    for (int dir = 0; dir < 2; ++dir) {
      int w = layer * 2 + dir;
      gemmLDS(apScr, pIh[w], nullptr, preBuf, SS * BB, H4, layerK[layer], 0);
      zero(hSt, BB * HH); zero(cSt, BB * HH);
      for (int step = 0; step < SS; ++step) {
        int s = dir ? (SS - 1 - step) : step;
        gemmF(hSt, pHh[w], nullptr, hg0, BB, H4, HH, 0);
        lstm_cell<<<(BB * HH + 255) / 256, 256, 0, stream>>>(
            preBuf + (size_t)s * BB * H4, hg0, bE[w], hSt, cSt,
            layerOut[layer] + (size_t)s * BB * 1024 + dir * HH, 1024);
      }
    }
  }

  // ---- 4. attention scale projection + static packed context half ----
  packA(encbuf, apScr, SS * BB, 1024);
  gemmLDS(apScr, pSc, bsc, scaleB, SS * BB, HH, 1024, 0);
  {
    size_t n = (size_t)BB * SS * 1024;
    ctx_fill_enc_packed<<<(unsigned)((n + 255) / 256), 256, 0, stream>>>(encbuf, ctxAp);
  }

  // ---- 5. decoder layer-0 input projection for all steps (padded M) ----
  packA(xdec, apScr, MDEC, EE);
  gemmLDS(apScr, pDih0, nullptr, preBuf, MDEC, H4, EE, 0);

  // ---- 6. decoder loop ----
  zero(h0St, BB * HH); zero(c0St, BB * HH);
  zero(h1St, BB * HH); zero(c1St, BB * HH);
  const size_t emisBase = (size_t)BB * SS + (size_t)(TT - 2) * BB * SS * SS;
  const size_t emisStep = (size_t)BB * SS * VV;
  for (int t = 0; t < TT - 1; ++t) {
    gemmF(h0St, pDhh0, nullptr, hg0, BB, H4, HH, 0);
    lstm_cell<<<(BB * HH + 255) / 256, 256, 0, stream>>>(
        preBuf + (size_t)t * BB * H4, hg0, dB0, h0St, c0St, nullptr, 0);
    gemmF(h0St, pDih1, nullptr, xg1, BB, H4, HH, 0);
    gemmF(h1St, pDhh1, nullptr, hg1, BB, H4, HH, 0);
    lstm_cell<<<(BB * HH + 255) / 256, 256, 0, stream>>>(
        xg1, hg1, dB1, h1St, c1St, nullptr, 0);

    attn_score<<<(BB * SS + 255) / 256, 256, 0, stream>>>(scaleB, h1St, score);
    attn_norm<<<BB, 32, 0, stream>>>(score, src_mask, ltr + (size_t)t * BB * SS);

    {
      size_t n = (size_t)BB * SS * HH;
      ctx_fill_h_packed<<<(unsigned)((n + 255) / 256), 256, 0, stream>>>(h1St, ctxAp);
    }
    gemmLDS(ctxAp, pLo, blo, ctx1, BB * SS, OD, OD, 1);       // tanh
    packA(ctx1, apScr, BB * SS, OD);
    gemmLDS(apScr, pFo, bfo, logits, BB * SS, VV, OD, 0);
    log_softmax_rows<<<(BB * SS + 7) / 8, 256, 0, stream>>>(
        logits, out + emisBase + (size_t)t * emisStep, BB * SS);
  }

  // ---- 7. assemble initial + transition outputs ----
  copy_initial<<<(BB * SS + 255) / 256, 256, 0, stream>>>(ltr, out);
  {
    size_t total = (size_t)(TT - 2) * BB * SS * SS;
    transition_bcast<<<(unsigned)((total + 255) / 256), 256, 0, stream>>>(ltr, out + BB * SS);
  }
}
